// ViTRoPEEncoder_24111946400309
// MI455X (gfx1250) — matrix operand order repaired and checked by reading, not compiled, not run on hardware
//
#include <hip/hip_runtime.h>

// ---------------------------------------------------------------------------
// Types for CDNA5 WMMA (wave32): v_wmma_f32_16x16x32_bf16
// ---------------------------------------------------------------------------
typedef __bf16 bf16;
typedef __attribute__((ext_vector_type(16))) bf16  v16bf;
typedef __attribute__((ext_vector_type(8)))  bf16  v8bf;
typedef __attribute__((ext_vector_type(8)))  float v8f;

#define S_TOT   2114          // 1 + 2*512 + 1 + 2*512 + 64
#define S_PAD   2144          // padded to multiple of 32 (16B-aligned rows)
#define EMB     384
#define NB      8
#define NH      8
#define DH      48
#define DFF     1536
#define MROWS   (NB * S_TOT)  // 16912 = 1057*16 = 2114*8
#define LEN_M1  1025
#define T_END   2050

__device__ __forceinline__ unsigned short f2bfraw(float f) {
  unsigned u = __builtin_bit_cast(unsigned, f);
  unsigned r = u + 0x7FFFu + ((u >> 16) & 1u);   // round-to-nearest-even
  return (unsigned short)(r >> 16);
}
__device__ __forceinline__ bf16 f2bf(float f) {
  unsigned short h = f2bfraw(f);
  return __builtin_bit_cast(bf16, h);
}
__device__ __forceinline__ unsigned packbf2(float f0, float f1) {
  return ((unsigned)f2bfraw(f1) << 16) | (unsigned)f2bfraw(f0);
}

__device__ __forceinline__ v16bf cat8(v8bf lo, v8bf hi) {
  v16bf r;
#pragma unroll
  for (int i = 0; i < 8; ++i) { r[i] = lo[i]; r[8 + i] = hi[i]; }
  return r;
}

// ---------------------------------------------------------------------------
// Input concatenation: x[B,S,384]
// ---------------------------------------------------------------------------
__global__ __launch_bounds__(256) void k_concat_x(
    const float* __restrict__ hand_t, const float* __restrict__ head_t,
    const float* __restrict__ hand_m1, const float* __restrict__ head_m1,
    const float* __restrict__ state_t, const float* __restrict__ state_m1,
    const float* __restrict__ perc, float* __restrict__ x)
{
  size_t idx = (size_t)blockIdx.x * 256 + threadIdx.x;
  size_t total = (size_t)MROWS * EMB;
  if (idx >= total) return;
  int d = (int)(idx % EMB);
  size_t rs = idx / EMB;
  int b = (int)(rs / S_TOT), s = (int)(rs % S_TOT);
  float v;
  if (s == 0)          v = state_m1[(size_t)b * EMB + d];
  else if (s < 513)    v = hand_m1[((size_t)b * 512 + (s - 1)) * EMB + d];
  else if (s < 1025)   v = head_m1[((size_t)b * 512 + (s - 513)) * EMB + d];
  else if (s == 1025)  v = state_t[(size_t)b * EMB + d];
  else if (s < 1538)   v = hand_t[((size_t)b * 512 + (s - 1026)) * EMB + d];
  else if (s < 2050)   v = head_t[((size_t)b * 512 + (s - 1538)) * EMB + d];
  else                 v = perc[((size_t)b * 64 + (s - 2050)) * EMB + d];
  x[idx] = v;
}

__global__ __launch_bounds__(256) void k_coords(
    const float* __restrict__ chand_t, const float* __restrict__ chead_t,
    const float* __restrict__ chand_m1, const float* __restrict__ chead_m1,
    float* __restrict__ coords)
{
  size_t idx = (size_t)blockIdx.x * 256 + threadIdx.x;
  size_t total = (size_t)MROWS * 3;
  if (idx >= total) return;
  int c = (int)(idx % 3);
  size_t rs = idx / 3;
  int b = (int)(rs / S_TOT), s = (int)(rs % S_TOT);
  float v = 0.f;
  if (s >= 1 && s < 513)          v = chand_m1[((size_t)b * 512 + (s - 1)) * 3 + c];
  else if (s >= 513 && s < 1025)  v = chead_m1[((size_t)b * 512 + (s - 513)) * 3 + c];
  else if (s >= 1026 && s < 1538) v = chand_t[((size_t)b * 512 + (s - 1026)) * 3 + c];
  else if (s >= 1538 && s < 2050) v = chead_t[((size_t)b * 512 + (s - 1538)) * 3 + c];
  coords[idx] = v;
}

// Weight convert+transpose: W[K][N] fp32 -> Wt[N][K] bf16 (K contiguous)
__global__ __launch_bounds__(256) void k_wt(const float* __restrict__ w,
                                            bf16* __restrict__ wt, int K, int N)
{
  size_t idx = (size_t)blockIdx.x * 256 + threadIdx.x;
  if (idx >= (size_t)K * N) return;
  int n = (int)(idx % N), k = (int)(idx / N);
  wt[(size_t)n * K + k] = f2bf(w[idx]);
}

// ---------------------------------------------------------------------------
// LayerNorm: one wave per 384-wide row (wave32 shuffle reduction)
// ---------------------------------------------------------------------------
__global__ __launch_bounds__(256) void k_ln(const float* __restrict__ x,
                                            const float* __restrict__ gg,
                                            const float* __restrict__ bb,
                                            float* __restrict__ out)
{
  int wv = threadIdx.x >> 5, lane = threadIdx.x & 31;
  int row = blockIdx.x * 8 + wv;
  if (row >= MROWS) return;
  const float* xr = x + (size_t)row * EMB;
  float s = 0.f, s2 = 0.f;
#pragma unroll
  for (int i = 0; i < EMB / 32; ++i) {
    float v = xr[lane + i * 32]; s += v; s2 += v * v;
  }
#pragma unroll
  for (int m = 1; m < 32; m <<= 1) { s += __shfl_xor(s, m, 32); s2 += __shfl_xor(s2, m, 32); }
  float mu = s * (1.f / EMB);
  float var = s2 * (1.f / EMB) - mu * mu;
  float rs = rsqrtf(var + 1e-6f);
  float* orow = out + (size_t)row * EMB;
#pragma unroll
  for (int i = 0; i < EMB / 32; ++i) {
    int c = lane + i * 32;
    orow[c] = (xr[c] - mu) * rs * gg[c] + bb[c];
  }
}

// ---------------------------------------------------------------------------
// GEMM: out[M,N] = act(A[M,K] @ W[K,N] + bias) (+ residual)
// A fp32, staged ONCE per block as a full 16 x K bf16 strip in LDS (no
// barriers inside the K loop).  W transposed bf16 Wt[N][K].
// Block: 256 thr = 8 waves; block tile 16 rows x 384 cols; wave tile 16x48
// (3 accumulators -> A-frag reused across 3 WMMAs).
// ---------------------------------------------------------------------------
template <int K>
__global__ __launch_bounds__(256) void k_gemm(
    const float* __restrict__ A, const bf16* __restrict__ Wt,
    const float* __restrict__ bias, const float* res,
    float* out, int N, int do_gelu)
{
  constexpr int STR = K + 8;                 // halves; 2*STR multiple of 16B
  __shared__ bf16 lsa[16][STR];
  int wv = threadIdx.x >> 5, lane = threadIdx.x & 31;
  int g = lane >> 4, n16 = lane & 15;
  int rowbase = blockIdx.y * 16;
  int colbase = blockIdx.x * 384 + wv * 48;

  // stage full 16 x K strip, vectorized: float4 load -> 4xbf16 ds_store_b64
  for (int i = threadIdx.x; i < (16 * K) / 4; i += 256) {
    int row = (i * 4) / K, col = (i * 4) % K;
    const float4 f = *(const float4*)&A[(size_t)(rowbase + row) * K + col];
    uint2 p; p.x = packbf2(f.x, f.y); p.y = packbf2(f.z, f.w);
    *(uint2*)&lsa[row][col] = p;
  }
  __syncthreads();

  v8f acc[3] = {{}, {}, {}};
  for (int kk = 0; kk < K; kk += 32) {
    // A frag: lane row = n16; K runs [kk+g*8, +8) and [kk+16+g*8, +8)
    const bf16* ap = &lsa[n16][kk + g * 8];
    v16bf av = cat8(*(const v8bf*)ap, *(const v8bf*)(ap + 16));
#pragma unroll
    for (int nt = 0; nt < 3; ++nt) {
      // B frag: lane col = n16; K runs [kk+g*8, +8) and [kk+16+g*8, +8) from Wt[N][K]
      const bf16* wp = Wt + (size_t)(colbase + nt * 16 + n16) * K + kk + g * 8;
      v16bf bv = cat8(*(const v8bf*)wp, *(const v8bf*)(wp + 16));
      if (kk + 32 < K) __builtin_prefetch((const void*)(wp + 32), 0, 1);
      acc[nt] = __builtin_amdgcn_wmma_f32_16x16x32_bf16(false, av, false, bv,
                                                        (short)0, acc[nt], false, false);
    }
  }
#pragma unroll
  for (int nt = 0; nt < 3; ++nt) {
    int col = colbase + nt * 16 + n16;
    float bcol = bias[col];
#pragma unroll
    for (int v = 0; v < 8; ++v) {
      int row = rowbase + v + 8 * g;
      float c = acc[nt][v] + bcol;
      if (do_gelu) c = 0.5f * c * (1.f + erff(c * 0.70710678118654752f));
      if (res) c += res[(size_t)row * N + col];
      out[(size_t)row * N + col] = c;
    }
  }
}

// ---------------------------------------------------------------------------
// RoPE + head reformat.  qkv[B,S,1152] fp32 ->
//   qro/kro[B,H,S_PAD,64] bf16 (d 48..63 zero), vT[B,H,48,S_PAD] bf16.
// ---------------------------------------------------------------------------
__global__ __launch_bounds__(192) void k_rope(const float* __restrict__ qkv,
                                              const float* __restrict__ coords,
                                              bf16* __restrict__ qro,
                                              bf16* __restrict__ kro,
                                              bf16* __restrict__ vT)
{
  int bs = blockIdx.x;
  int b = bs / S_TOT, s = bs % S_TOT;
  int t = threadIdx.x;
  const float* base = qkv + (size_t)bs * 1152;
  {
    int h = t / 24, pp = t % 24, ax = pp / 8, p = pp % 8;
    float inv = powf(10000.f, -(float)p * 0.125f);
    float ang = coords[(size_t)bs * 3 + ax] * inv;
    float c = cosf(ang), sn = sinf(ang);
    int d0 = 2 * pp;
    size_t ro = ((size_t)(b * NH + h) * S_PAD + s) * 64;
    float q0 = base[h * DH + d0], q1 = base[h * DH + d0 + 1];
    qro[ro + d0]     = f2bf(q0 * c - q1 * sn);
    qro[ro + d0 + 1] = f2bf(q0 * sn + q1 * c);
    float k0 = base[EMB + h * DH + d0], k1 = base[EMB + h * DH + d0 + 1];
    kro[ro + d0]     = f2bf(k0 * c - k1 * sn);
    kro[ro + d0 + 1] = f2bf(k0 * sn + k1 * c);
  }
  for (int i = t; i < NH * 16; i += 192) {      // zero-pad d = 48..63
    int h = i / 16, d = DH + (i % 16);
    size_t ro = ((size_t)(b * NH + h) * S_PAD + s) * 64;
    qro[ro + d] = f2bf(0.f);
    kro[ro + d] = f2bf(0.f);
  }
  for (int i = t; i < NH * DH; i += 192) {      // V transposed
    int h = i / DH, d = i % DH;
    vT[((size_t)(b * NH + h) * DH + d) * S_PAD + s] = f2bf(base[2 * EMB + h * DH + d]);
  }
}

// ---------------------------------------------------------------------------
// Flash attention: one wave per (b, h, 32-query group = 2 x 16-row tiles).
// K/V fragments are loaded once per 32-key tile and reused by both query
// tiles (2x arithmetic intensity).  Mask predicates are hoisted to a uniform
// per-tile branch so the common path is a bare scale.
// ---------------------------------------------------------------------------
__global__ __launch_bounds__(64) void k_attn(const bf16* __restrict__ qro,
                                             const bf16* __restrict__ kro,
                                             const bf16* __restrict__ vT,
                                             float* __restrict__ out)
{
  __shared__ bf16 lp[2][2][16][32];             // [wave][qtile] P tiles
  int wv = threadIdx.x >> 5, lane = threadIdx.x & 31;
  int grp = blockIdx.x * 2 + wv;
  if (grp >= 67) return;                        // ceil(2114/32) groups of 32 rows
  int h = blockIdx.y, b = blockIdx.z;
  int bh = b * NH + h;
  int g = lane >> 4, n16 = lane & 15;
  int qbase = grp * 32;                         // rows [qbase, qbase+32) < S_PAD

  v16bf qa[2][2];
#pragma unroll
  for (int qt = 0; qt < 2; ++qt)
#pragma unroll
    for (int st = 0; st < 2; ++st) {
      const bf16* qp = qro + ((size_t)bh * S_PAD + qbase + qt * 16 + n16) * 64
                           + st * 32 + g * 8;
      qa[qt][st] = cat8(*(const v8bf*)qp, *(const v8bf*)(qp + 16));
    }

  v8f zz = {};
  v8f o[2][3];
  float mrow[2][8], lrow[2][8];
#pragma unroll
  for (int qt = 0; qt < 2; ++qt) {
#pragma unroll
    for (int nt = 0; nt < 3; ++nt) o[qt][nt] = zz;
#pragma unroll
    for (int v = 0; v < 8; ++v) { mrow[qt][v] = -1e30f; lrow[qt][v] = 0.f; }
  }
  const float SCALE = 0.14433756729740643f;     // 1/sqrt(48)

  for (int kb = 0; kb < S_PAD; kb += 32) {
    // --- K fragments, shared by both query tiles ---
    v16bf kv[2][2];
#pragma unroll
    for (int hf = 0; hf < 2; ++hf)
#pragma unroll
      for (int st = 0; st < 2; ++st) {
        const bf16* kp = kro + ((size_t)bh * S_PAD + kb + hf * 16 + n16) * 64
                             + st * 32 + g * 8;
        kv[hf][st] = cat8(*(const v8bf*)kp, *(const v8bf*)(kp + 16));
      }
    // --- scores ---
    float sc[2][2][8];
#pragma unroll
    for (int qt = 0; qt < 2; ++qt)
#pragma unroll
      for (int hf = 0; hf < 2; ++hf) {
        v8f s = zz;
#pragma unroll
        for (int st = 0; st < 2; ++st)
          s = __builtin_amdgcn_wmma_f32_16x16x32_bf16(false, qa[qt][st], false,
                                                      kv[hf][st], (short)0, s,
                                                      false, false);
#pragma unroll
        for (int v = 0; v < 8; ++v) sc[qt][hf][v] = s[v];
      }
    // --- scale + (rare) mask: uniform predicate -> scalar branch ---
    bool domask = (kb + 32 > S_TOT) ||
                  (qbase < LEN_M1 && (kb + 32 > LEN_M1) && (kb < T_END));
    if (domask) {
#pragma unroll
      for (int qt = 0; qt < 2; ++qt)
#pragma unroll
        for (int hf = 0; hf < 2; ++hf) {
          int kcol = kb + hf * 16 + n16;
#pragma unroll
          for (int v = 0; v < 8; ++v) {
            int qrow = qbase + qt * 16 + v + 8 * g;
            float x = sc[qt][hf][v] * SCALE;
            if (kcol >= S_TOT ||
                (qrow < LEN_M1 && kcol >= LEN_M1 && kcol < T_END))
              x = -1e30f;                        // finite "masked": exp -> 0
            sc[qt][hf][v] = x;
          }
        }
    } else {
#pragma unroll
      for (int qt = 0; qt < 2; ++qt)
#pragma unroll
        for (int hf = 0; hf < 2; ++hf)
#pragma unroll
          for (int v = 0; v < 8; ++v) sc[qt][hf][v] *= SCALE;
    }
    // --- online softmax (per query tile) ---
#pragma unroll
    for (int qt = 0; qt < 2; ++qt) {
      float al[8];
#pragma unroll
      for (int v = 0; v < 8; ++v) {
        float rm = fmaxf(sc[qt][0][v], sc[qt][1][v]);
#pragma unroll
        for (int m = 1; m < 16; m <<= 1) rm = fmaxf(rm, __shfl_xor(rm, m, 32));
        float mnew = fmaxf(mrow[qt][v], rm);
        al[v] = expf(mrow[qt][v] - mnew);
        mrow[qt][v] = mnew;
        float p0 = expf(sc[qt][0][v] - mnew);
        float p1 = expf(sc[qt][1][v] - mnew);
        lp[wv][qt][v + 8 * g][n16]      = f2bf(p0);
        lp[wv][qt][v + 8 * g][16 + n16] = f2bf(p1);
        float su = p0 + p1;
#pragma unroll
        for (int m = 1; m < 16; m <<= 1) su += __shfl_xor(su, m, 32);
        lrow[qt][v] = lrow[qt][v] * al[v] + su;
      }
#pragma unroll
      for (int v = 0; v < 8; ++v) {
        o[qt][0][v] *= al[v]; o[qt][1][v] *= al[v]; o[qt][2][v] *= al[v];
      }
    }
    asm volatile("s_wait_dscnt 0" ::: "memory"); // P stores visible to A-frag loads
    // --- V fragments, shared by both query tiles ---
    v16bf vv[3];
#pragma unroll
    for (int nt = 0; nt < 3; ++nt) {
      const bf16* vp = vT + ((size_t)bh * DH + nt * 16 + n16) * S_PAD + kb + g * 8;
      vv[nt] = cat8(*(const v8bf*)vp, *(const v8bf*)(vp + 16));
    }
#pragma unroll
    for (int qt = 0; qt < 2; ++qt) {
      const bf16* pr = &lp[wv][qt][n16][g * 8];
      v16bf pa = cat8(*(const v8bf*)pr, *(const v8bf*)(pr + 16));
#pragma unroll
      for (int nt = 0; nt < 3; ++nt)
        o[qt][nt] = __builtin_amdgcn_wmma_f32_16x16x32_bf16(false, pa, false,
                                                            vv[nt], (short)0,
                                                            o[qt][nt], false, false);
    }
  }
#pragma unroll
  for (int qt = 0; qt < 2; ++qt)
#pragma unroll
    for (int v = 0; v < 8; ++v) {
      int row = qbase + qt * 16 + v + 8 * g;
      if (row >= S_TOT) continue;
      float inv = 1.f / lrow[qt][v];
      size_t oo = ((size_t)b * S_TOT + row) * EMB + h * DH;
      out[oo + n16]      = o[qt][0][v] * inv;
      out[oo + 16 + n16] = o[qt][1][v] * inv;
      out[oo + 32 + n16] = o[qt][2][v] * inv;
    }
}

__global__ __launch_bounds__(256) void k_slice(const float* __restrict__ xin,
                                               float* __restrict__ out)
{
  size_t idx = (size_t)blockIdx.x * 256 + threadIdx.x;
  size_t total = (size_t)NB * 1088 * EMB;
  if (idx >= total) return;
  int d = (int)(idx % EMB);
  size_t rs = idx / EMB;
  int b = (int)(rs / 1088), j = (int)(rs % 1088);
  out[idx] = xin[((size_t)b * S_TOT + 1026 + j) * EMB + d];
}

// ---------------------------------------------------------------------------
// Host orchestration (launch-only: graph-capture safe)
// ---------------------------------------------------------------------------
extern "C" void kernel_launch(void* const* d_in, const int* in_sizes, int n_in,
                              void* d_out, int out_size, void* d_ws, size_t ws_size,
                              hipStream_t stream)
{
  (void)in_sizes; (void)n_in; (void)out_size; (void)ws_size;
  const float* hand_t   = (const float*)d_in[0];
  const float* head_t   = (const float*)d_in[1];
  const float* hand_m1  = (const float*)d_in[2];
  const float* head_m1  = (const float*)d_in[3];
  const float* chand_t  = (const float*)d_in[4];
  const float* chead_t  = (const float*)d_in[5];
  const float* chand_m1 = (const float*)d_in[6];
  const float* chead_m1 = (const float*)d_in[7];
  const float* state_t  = (const float*)d_in[8];
  const float* state_m1 = (const float*)d_in[9];
  const float* perc     = (const float*)d_in[10];
  auto P = [&](int l, int i) { return (const float*)d_in[11 + l * 12 + i]; };
  const float* norm_g = (const float*)d_in[59];
  const float* norm_b = (const float*)d_in[60];

  char* ws = (char*)d_ws;
  size_t off = 0;
  auto alloc = [&](size_t bytes) -> char* {
    size_t cur = off;
    off += (bytes + 255) & ~(size_t)255;
    return ws + cur;
  };
  float* xbuf   = (float*)alloc((size_t)MROWS * EMB * 4);
  float* habuf  = (float*)alloc((size_t)MROWS * EMB * 4);   // h / attn_out / h2
  float* bigbuf = (float*)alloc((size_t)MROWS * DFF * 4);   // qkv / mlp hidden
  float* coords = (float*)alloc((size_t)MROWS * 3 * 4);
  bf16* qro = (bf16*)alloc((size_t)NB * NH * S_PAD * 64 * 2);
  bf16* kro = (bf16*)alloc((size_t)NB * NH * S_PAD * 64 * 2);
  bf16* vTb = (bf16*)alloc((size_t)NB * NH * DH * S_PAD * 2);
  bf16* wts[4][4];
  const int wk[4] = {EMB, EMB, EMB, DFF};
  const int wn[4] = {3 * EMB, EMB, DFF, EMB};
  const int widx[4] = {2, 4, 8, 10};            // qkv_w, proj_w, fc1_w, fc2_w
  for (int l = 0; l < 4; ++l)
    for (int w = 0; w < 4; ++w)
      wts[l][w] = (bf16*)alloc((size_t)wk[w] * wn[w] * 2);

  // weight convert+transpose (once per launch)
  for (int l = 0; l < 4; ++l)
    for (int w = 0; w < 4; ++w) {
      size_t tot = (size_t)wk[w] * wn[w];
      k_wt<<<(unsigned)((tot + 255) / 256), 256, 0, stream>>>(P(l, widx[w]),
                                                              wts[l][w], wk[w], wn[w]);
    }

  size_t totx = (size_t)MROWS * EMB;
  k_concat_x<<<(unsigned)((totx + 255) / 256), 256, 0, stream>>>(
      hand_t, head_t, hand_m1, head_m1, state_t, state_m1, perc, xbuf);
  size_t totc = (size_t)MROWS * 3;
  k_coords<<<(unsigned)((totc + 255) / 256), 256, 0, stream>>>(
      chand_t, chead_t, chand_m1, chead_m1, coords);

  for (int l = 0; l < 4; ++l) {
    k_ln<<<MROWS / 8, 256, 0, stream>>>(xbuf, P(l, 0), P(l, 1), habuf);
    k_gemm<384><<<dim3(1152 / 384, MROWS / 16), 256, 0, stream>>>(
        habuf, wts[l][0], P(l, 3), nullptr, bigbuf, 1152, 0);
    k_rope<<<MROWS, 192, 0, stream>>>(bigbuf, coords, qro, kro, vTb);
    k_attn<<<dim3(34, NH, NB), 64, 0, stream>>>(qro, kro, vTb, habuf);
    k_gemm<384><<<dim3(384 / 384, MROWS / 16), 256, 0, stream>>>(
        habuf, wts[l][1], P(l, 5), xbuf, xbuf, 384, 0);
    k_ln<<<MROWS / 8, 256, 0, stream>>>(xbuf, P(l, 6), P(l, 7), habuf);
    k_gemm<384><<<dim3(1536 / 384, MROWS / 16), 256, 0, stream>>>(
        habuf, wts[l][2], P(l, 9), nullptr, bigbuf, 1536, 1);
    k_gemm<1536><<<dim3(384 / 384, MROWS / 16), 256, 0, stream>>>(
        bigbuf, wts[l][3], P(l, 11), xbuf, xbuf, 384, 0);
  }
  k_ln<<<MROWS / 8, 256, 0, stream>>>(xbuf, norm_g, norm_b, habuf);
  size_t tots = (size_t)NB * 1088 * EMB;
  k_slice<<<(unsigned)((tots + 255) / 256), 256, 0, stream>>>(habuf, (float*)d_out);
}
